// GRU_48816598287016
// MI455X (gfx1250) — matrix operand order repaired and checked by reading, not compiled, not run on hardware
//
#include <hip/hip_runtime.h>
#include <hip/hip_bf16.h>
#include <math.h>

// ---------------------------------------------------------------------------
// 2-layer GRU LM forward for MI455X (gfx1250, wave32, WMMA bf16).
//
//  * Time-parallel GEMMs (layer-0 x@W1 projections; final vocab projection)
//    hoisted out of the sequential scan into two big WMMA GEMMs with
//    MSUB=2 m-tiles/wave (B-fragment reuse) and 2-stage software pipelining.
//  * Recurrence: 128 x 4 small WMMA kernels, now also software-pipelined
//    (the sequential scan is the latency-critical path; overlapping L2
//    weight-fragment loads with the WMMA pipe shortens each step's k-chain).
//    bf16 weights (~41 MB) are L2-resident (192 MB).
//  * Weights f32->bf16 once per launch; activations stay f32 and are
//    converted in-register when building A fragments (M=32, negligible).
// ---------------------------------------------------------------------------

typedef __bf16 bf16;
typedef __attribute__((ext_vector_type(16))) bf16  bf16x16;
typedef __attribute__((ext_vector_type(8)))  bf16  bf16x8;
typedef __attribute__((ext_vector_type(8)))  float f32x8;

#define EMB    512
#define HID    1024
#define SEQ    128
#define BATCH  32
#define VOCAB  10000
#define ROWS   (SEQ * BATCH)   // 4096 time*batch rows

#define WMMA_BF16(a, b, c) \
  __builtin_amdgcn_wmma_f32_16x16x32_bf16(false, (a), false, (b), (short)0, (c), false, false)

// ---- fragment loaders (layouts per CDNA5 ISA 7.12.2) -----------------------
// A 16x32 bf16: lane<16 holds row M=lane, K = k..k+7 (elems 0-7) and
// k+16..k+23 (elems 8-15); lane>=16 holds the +8 shifted K ranges.
__device__ __forceinline__ bf16x16 load_a_bf16(const bf16* A, int lda, int m0, int k, int lane) {
  int half = lane >> 4, r = lane & 15;
  const bf16* p = A + (size_t)(m0 + r) * lda + k + half * 8;
  bf16x8 lo = *(const bf16x8*)p;
  bf16x8 hi = *(const bf16x8*)(p + 16);
  bf16x16 a;
#pragma unroll
  for (int i = 0; i < 8; i++) { a[i] = lo[i]; a[i + 8] = hi[i]; }
  return a;
}

// A fragment from f32 activations, converted in-register. If MUL, the
// element is R[i]*A[i] (GRU reset-gated state r*h).
template <bool MUL>
__device__ __forceinline__ bf16x16 load_a_act(const float* A, const float* R, int lda,
                                              int m0, int k, int lane) {
  int half = lane >> 4, r = lane & 15;
  size_t base = (size_t)(m0 + r) * lda + k + half * 8;
  bf16x16 a;
#pragma unroll
  for (int i = 0; i < 8; i++) {
    float lo = A[base + i], hi = A[base + 16 + i];
    if (MUL) { lo *= R[base + i]; hi *= R[base + 16 + i]; }
    a[i] = (bf16)lo;
    a[i + 8] = (bf16)hi;
  }
  return a;
}

// B 32x16 bf16 from row-major weights W[N,K] (computing x @ W^T):
// lane<16: column n=lane -> W[n, k..k+7] and W[n, k+16..k+23];
// lane>=16: the +8 shifted K ranges (matches A).
__device__ __forceinline__ bf16x16 load_b_w(const bf16* W, int ldw, int n0, int k, int lane) {
  int half = lane >> 4, r = lane & 15;
  const bf16* p = W + (size_t)(n0 + r) * ldw + k + half * 8;
  bf16x8 lo = *(const bf16x8*)p;
  bf16x8 hi = *(const bf16x8*)(p + 16);
  bf16x16 b;
#pragma unroll
  for (int i = 0; i < 8; i++) { b[i] = lo[i]; b[i + 8] = hi[i]; }
  return b;
}

// ---- pipelined accumulate-GEMM for the recurrent steps ---------------------
// acc[0..3] += act[M16 tile] @ W[N,K]^T over 4 n-subtiles, K % 64 == 0.
// 2-stage static double buffer so fragment loads overlap the WMMA pipe.
template <bool MUL>
__device__ __forceinline__ void acc_gemm(f32x8 (&acc)[4], const float* A, const float* R,
                                         int K, const bf16* W, int m0, int n0, int lane) {
  bf16x16 a0, a1, b0[4], b1[4];
  a0 = load_a_act<MUL>(A, R, K, m0, 0, lane);
#pragma unroll
  for (int s = 0; s < 4; s++) b0[s] = load_b_w(W, K, n0 + 16 * s, 0, lane);
  for (int k = 0; k < K; k += 64) {
    if (k + 32 < K) {
      a1 = load_a_act<MUL>(A, R, K, m0, k + 32, lane);
#pragma unroll
      for (int s = 0; s < 4; s++) b1[s] = load_b_w(W, K, n0 + 16 * s, k + 32, lane);
    }
#pragma unroll
    for (int s = 0; s < 4; s++) acc[s] = WMMA_BF16(a0, b0[s], acc[s]);
    if (k + 64 < K) {
      a0 = load_a_act<MUL>(A, R, K, m0, k + 64, lane);
#pragma unroll
      for (int s = 0; s < 4; s++) b0[s] = load_b_w(W, K, n0 + 16 * s, k + 64, lane);
    }
#pragma unroll
    for (int s = 0; s < 4; s++) acc[s] = WMMA_BF16(a1, b1[s], acc[s]);
  }
}

// ---- big time-parallel GEMM: C[M,N] = A[M,K](bf16) @ W[N,K]^T (+bias) ------
// MSUB m-tiles per wave reuse each B fragment; 2-stage pipeline; K % 64 == 0.
template <int MSUB, int NSUB>
__global__ __launch_bounds__(128) void gemm_bf16(const bf16* __restrict__ A,
                                                 const bf16* __restrict__ W,
                                                 const float* __restrict__ bias,
                                                 float* __restrict__ C,
                                                 int M, int N, int K) {
  int wave = blockIdx.x * (blockDim.x >> 5) + (threadIdx.x >> 5);
  int lane = threadIdx.x & 31;
  int ntiles = N / (16 * NSUB);
  int m0 = (wave / ntiles) * (16 * MSUB);
  int n0 = (wave % ntiles) * (16 * NSUB);
  if (m0 >= M) return;                 // wave-uniform: EXEC stays all-ones
  f32x8 acc[MSUB][NSUB] = {};
  bf16x16 a0[MSUB], a1[MSUB], b0[NSUB], b1[NSUB];

#pragma unroll
  for (int mi = 0; mi < MSUB; mi++) a0[mi] = load_a_bf16(A, K, m0 + 16 * mi, 0, lane);
#pragma unroll
  for (int s = 0; s < NSUB; s++)    b0[s]  = load_b_w(W, K, n0 + 16 * s, 0, lane);

  for (int k = 0; k < K; k += 64) {
    // prefetch k+32 while computing with k
    if (k + 32 < K) {
#pragma unroll
      for (int mi = 0; mi < MSUB; mi++) a1[mi] = load_a_bf16(A, K, m0 + 16 * mi, k + 32, lane);
#pragma unroll
      for (int s = 0; s < NSUB; s++)    b1[s]  = load_b_w(W, K, n0 + 16 * s, k + 32, lane);
    }
#pragma unroll
    for (int mi = 0; mi < MSUB; mi++)
#pragma unroll
      for (int s = 0; s < NSUB; s++)
        acc[mi][s] = WMMA_BF16(a0[mi], b0[s], acc[mi][s]);
    // prefetch k+64 while computing with k+32
    if (k + 64 < K) {
#pragma unroll
      for (int mi = 0; mi < MSUB; mi++) a0[mi] = load_a_bf16(A, K, m0 + 16 * mi, k + 64, lane);
#pragma unroll
      for (int s = 0; s < NSUB; s++)    b0[s]  = load_b_w(W, K, n0 + 16 * s, k + 64, lane);
    }
#pragma unroll
    for (int mi = 0; mi < MSUB; mi++)
#pragma unroll
      for (int s = 0; s < NSUB; s++)
        acc[mi][s] = WMMA_BF16(a1[mi], b1[s], acc[mi][s]);
  }

  int half = lane >> 4, r = lane & 15;
#pragma unroll
  for (int mi = 0; mi < MSUB; mi++) {
#pragma unroll
    for (int s = 0; s < NSUB; s++) {
      int col = n0 + 16 * s + r;
      float bb = bias ? bias[col] : 0.0f;
#pragma unroll
      for (int g = 0; g < 8; g++) {    // C/D layout: VGPR g -> M=g / g+8
        int row = m0 + 16 * mi + half * 8 + g;
        C[(size_t)row * N + col] = acc[mi][s][g] + bb;
      }
    }
  }
}

// ---- recurrent step, phase 1: r,z = sigmoid([x@W1] + h@W2 + pre + b) -------
// N = 2*HID (r|z concatenated). pre = precomputed x@W1 (layer 0) or null.
__global__ __launch_bounds__(128) void gru_rz(const float* __restrict__ x,
                                              const bf16* __restrict__ W1, int Kx,
                                              const float* __restrict__ h,
                                              const bf16* __restrict__ W2,
                                              const float* __restrict__ pre,
                                              const float* __restrict__ br,
                                              const float* __restrict__ bz,
                                              float* __restrict__ rbuf,
                                              float* __restrict__ zbuf) {
  const int N = 2 * HID;
  int wave = blockIdx.x * (blockDim.x >> 5) + (threadIdx.x >> 5);
  int lane = threadIdx.x & 31;
  const int ntiles = N / 64;
  int m0 = (wave / ntiles) * 16;
  int n0 = (wave % ntiles) * 64;
  if (m0 >= BATCH) return;
  f32x8 acc[4] = {};
  if (x) acc_gemm<false>(acc, x, nullptr, Kx, W1, m0, n0, lane);
  acc_gemm<false>(acc, h, nullptr, HID, W2, m0, n0, lane);

  int half = lane >> 4, r = lane & 15;
#pragma unroll
  for (int s = 0; s < 4; s++) {
    int col = n0 + 16 * s + r;
#pragma unroll
    for (int g = 0; g < 8; g++) {
      int row = m0 + half * 8 + g;
      float v = acc[s][g];
      if (pre) v += pre[(size_t)row * (3 * HID) + col];
      v += (col < HID) ? br[col] : bz[col - HID];
      float sg = 1.0f / (1.0f + __expf(-v));
      if (col < HID) rbuf[row * HID + col] = sg;
      else           zbuf[row * HID + col - HID] = sg;
    }
  }
}

// ---- recurrent step, phase 2: h' = (1-z)*h + z*tanh([x@W1h] + (r*h)@W2h +…)
__global__ __launch_bounds__(128) void gru_h(const float* __restrict__ x,
                                             const bf16* __restrict__ W1h, int Kx,
                                             const float* __restrict__ r,
                                             const float* __restrict__ h,
                                             const bf16* __restrict__ W2h,
                                             const float* __restrict__ pre,
                                             const float* __restrict__ bh,
                                             const float* __restrict__ z,
                                             float* __restrict__ hnew,
                                             bf16* __restrict__ hbf) {
  int wave = blockIdx.x * (blockDim.x >> 5) + (threadIdx.x >> 5);
  int lane = threadIdx.x & 31;
  const int ntiles = HID / 64;
  int m0 = (wave / ntiles) * 16;
  int n0 = (wave % ntiles) * 64;
  if (m0 >= BATCH) return;
  f32x8 acc[4] = {};
  if (x) acc_gemm<false>(acc, x, nullptr, Kx, W1h, m0, n0, lane);
  acc_gemm<true>(acc, h, r, HID, W2h, m0, n0, lane);   // A = r*h, fused cvt

  int half = lane >> 4, rr = lane & 15;
#pragma unroll
  for (int s = 0; s < 4; s++) {
    int col = n0 + 16 * s + rr;
#pragma unroll
    for (int g = 0; g < 8; g++) {
      int row = m0 + half * 8 + g;
      float v = acc[s][g];
      if (pre) v += pre[(size_t)row * (3 * HID) + col];
      v += bh[col];
      float ht = tanhf(v);
      float zz = z[row * HID + col];
      float hn = (1.0f - zz) * h[row * HID + col] + zz * ht;
      hnew[row * HID + col] = hn;
      if (hbf) hbf[row * HID + col] = (bf16)hn;
    }
  }
}

// ---- small utility kernels -------------------------------------------------
__global__ void f32_to_bf16(const float* __restrict__ src, bf16* __restrict__ dst, int n) {
  int i = blockIdx.x * blockDim.x + threadIdx.x;
  if (i < n) dst[i] = (bf16)src[i];
}

__global__ void embed_bf16(const int* __restrict__ tok, const float* __restrict__ emb,
                           bf16* __restrict__ X) {
  int i = blockIdx.x * blockDim.x + threadIdx.x;   // ROWS*EMB threads
  if (i >= ROWS * EMB) return;
  int row = i / EMB, c = i % EMB;
  X[i] = (bf16)emb[(size_t)tok[row] * EMB + c];
}

// ---------------------------------------------------------------------------
extern "C" void kernel_launch(void* const* d_in, const int* in_sizes, int n_in,
                              void* d_out, int out_size, void* d_ws, size_t ws_size,
                              hipStream_t stream) {
  (void)in_sizes; (void)n_in; (void)out_size; (void)ws_size;
  // Input order = setup_inputs() insertion order, depth-first:
  // inputs, hidden, emb, then per layer (r,z,h) each (W1,W2,b2), out_W, out_b.
  const int*   tok  = (const int*)d_in[0];
  const float* hid0 = (const float*)d_in[1];
  const float* emb  = (const float*)d_in[2];
  const float* W1p[2][3]; const float* W2p[2][3]; const float* B2p[2][3];
  int idx = 3;
  for (int l = 0; l < 2; l++)
    for (int g = 0; g < 3; g++) {          // g: 0=r, 1=z, 2=h
      W1p[l][g] = (const float*)d_in[idx++];
      W2p[l][g] = (const float*)d_in[idx++];
      B2p[l][g] = (const float*)d_in[idx++];
    }
  const float* outW = (const float*)d_in[idx++];
  const float* outB = (const float*)d_in[idx++];

  // ---- workspace bump allocator (all buffers fully written before read) ---
  char* base = (char*)d_ws;
  size_t off = 0;
  auto alloc = [&](size_t bytes) -> void* {
    off = (off + 255) & ~(size_t)255;
    void* p = base + off;
    off += bytes;
    return p;
  };
  bf16*  Xbf  = (bf16*)alloc((size_t)ROWS * EMB * 2);
  bf16*  W1l0 = (bf16*)alloc((size_t)3 * HID * EMB * 2);   // r|z|h stacked
  bf16*  W2l0 = (bf16*)alloc((size_t)3 * HID * HID * 2);
  bf16*  W1l1 = (bf16*)alloc((size_t)3 * HID * HID * 2);
  bf16*  W2l1 = (bf16*)alloc((size_t)3 * HID * HID * 2);
  bf16*  OWbf = (bf16*)alloc((size_t)VOCAB * HID * 2);
  float* pre0 = (float*)alloc((size_t)ROWS * 3 * HID * 4); // x@W1 for layer 0
  bf16*  H1bf = (bf16*)alloc((size_t)ROWS * HID * 2);      // layer-1 h, all t
  float* h0b[2] = { (float*)alloc(BATCH * HID * 4), (float*)alloc(BATCH * HID * 4) };
  float* h1b[2] = { (float*)alloc(BATCH * HID * 4), (float*)alloc(BATCH * HID * 4) };
  float* r0 = (float*)alloc(BATCH * HID * 4);
  float* z0 = (float*)alloc(BATCH * HID * 4);
  float* r1 = (float*)alloc(BATCH * HID * 4);
  float* z1 = (float*)alloc(BATCH * HID * 4);

  // ---- one-time (per launch) weight conversion to bf16 cat buffers --------
  auto cvt = [&](const float* s, bf16* d, size_t n) {
    f32_to_bf16<<<(int)((n + 255) / 256), 256, 0, stream>>>(s, d, (int)n);
  };
  for (int g = 0; g < 3; g++) {
    cvt(W1p[0][g], W1l0 + (size_t)g * HID * EMB, (size_t)HID * EMB);
    cvt(W2p[0][g], W2l0 + (size_t)g * HID * HID, (size_t)HID * HID);
    cvt(W1p[1][g], W1l1 + (size_t)g * HID * HID, (size_t)HID * HID);
    cvt(W2p[1][g], W2l1 + (size_t)g * HID * HID, (size_t)HID * HID);
  }
  cvt(outW, OWbf, (size_t)VOCAB * HID);

  // embedding gather for all timesteps
  embed_bf16<<<(ROWS * EMB) / 256, 256, 0, stream>>>(tok, emb, Xbf);

  // initial hidden state
  hipMemcpyAsync(h0b[0], hid0, (size_t)BATCH * HID * 4, hipMemcpyDeviceToDevice, stream);
  hipMemcpyAsync(h1b[0], hid0 + BATCH * HID, (size_t)BATCH * HID * 4,
                 hipMemcpyDeviceToDevice, stream);

  // big GEMM: all layer-0 input projections  [4096,512] x [512,3072]
  // MSUB=2,NSUB=4: waves = (4096/32)*(3072/64) = 6144 -> 1536 blocks
  gemm_bf16<2, 4><<<1536, 128, 0, stream>>>(Xbf, W1l0, nullptr, pre0, ROWS, 3 * HID, EMB);

  // ---- sequential recurrence ----------------------------------------------
  int p = 0;
  for (int t = 0; t < SEQ; t++) {
    const float* preT = pre0 + (size_t)t * BATCH * 3 * HID;
    // layer 0 (x-projection precomputed in pre0)
    gru_rz<<<16, 128, 0, stream>>>(nullptr, nullptr, 0, h0b[p], W2l0, preT,
                                   B2p[0][0], B2p[0][1], r0, z0);
    gru_h<<<8, 128, 0, stream>>>(nullptr, nullptr, 0, r0, h0b[p],
                                 W2l0 + (size_t)2 * HID * HID, preT + 2 * HID,
                                 B2p[0][2], z0, h0b[p ^ 1], nullptr);
    // layer 1 (x = new layer-0 hidden)
    gru_rz<<<16, 128, 0, stream>>>(h0b[p ^ 1], W1l1, HID, h1b[p], W2l1, nullptr,
                                   B2p[1][0], B2p[1][1], r1, z1);
    gru_h<<<8, 128, 0, stream>>>(h0b[p ^ 1], W1l1 + (size_t)2 * HID * HID, HID,
                                 r1, h1b[p], W2l1 + (size_t)2 * HID * HID, nullptr,
                                 B2p[1][2], z1, h1b[p ^ 1],
                                 H1bf + (size_t)t * BATCH * HID);
    p ^= 1;
  }

  // big GEMM: logits [4096,1024] x [1024,10000] (+out_b), straight to d_out.
  // MSUB=2,NSUB=5: n-tile 80, 10000/80 = 125; waves = 128*125 = 16000 -> 4000 blocks.
  gemm_bf16<2, 5><<<4000, 128, 0, stream>>>(H1bf, OWbf, outB, (float*)d_out,
                                            ROWS, VOCAB, HID);

  // final hidden state tail of d_out (p is back to 0 after 128 flips)
  float* outHid = (float*)d_out + (size_t)ROWS * VOCAB;
  hipMemcpyAsync(outHid, h0b[p], (size_t)BATCH * HID * 4, hipMemcpyDeviceToDevice, stream);
  hipMemcpyAsync(outHid + BATCH * HID, h1b[p], (size_t)BATCH * HID * 4,
                 hipMemcpyDeviceToDevice, stream);
}
